// BertSelfAttention_75170517614971
// MI455X (gfx1250) — compile-verified, matrix operand order checked by reading, not run on hardware
//
#include <hip/hip_runtime.h>
#include <hip/hip_bf16.h>

// ---------------- problem constants ----------------
#define Bc   4
#define Sc   1024
#define HIDc 768
#define Hc   12
#define Dc   64
#define BHc  (Bc * Hc)          // 48

typedef _Float16 v16h __attribute__((ext_vector_type(16)));
typedef _Float16 v8h  __attribute__((ext_vector_type(8)));
typedef _Float16 v4h  __attribute__((ext_vector_type(4)));
typedef float    v8f  __attribute__((ext_vector_type(8)));
typedef float    v4f  __attribute__((ext_vector_type(4)));

__device__ __forceinline__ v8f wmma_f16(v16h a, v16h b, v8f c) {
  // D = A(16x32,f16) * B(32x16,f16) + C(16x16,f32)
  return __builtin_amdgcn_wmma_f32_16x16x32_f16(
      /*neg_a=*/false, a, /*neg_b=*/false, b,
      /*c_mod=*/(short)0, c, /*reuse_a=*/false, /*reuse_b=*/false);
}

// Load one lane's share of a 16x32 f16 fragment (A or B operand).
// p must already point at: tile_base + (lane&15)*stride + ((lane&16)>>1)
__device__ __forceinline__ v16h ld_frag16(const _Float16* __restrict__ p) {
  union { v16h v; v8h h[2]; } u;
  u.h[0] = *(const v8h*)(p);        // K = kofs+0..7
  u.h[1] = *(const v8h*)(p + 16);   // K = kofs+16..23
  return u.v;
}

// ---------------- 1) f32 -> f16 convert ----------------
__global__ __launch_bounds__(256) void cvt_f32_f16_kernel(
    const float* __restrict__ in, _Float16* __restrict__ out, int n4) {
  int i = blockIdx.x * 256 + threadIdx.x;
  if (i < n4) {
    v4f x = ((const v4f*)in)[i];
    v4h y;
#pragma unroll
    for (int j = 0; j < 4; ++j) y[j] = (_Float16)x[j];
    ((v4h*)out)[i] = y;
  }
}

// ---------------- 2) QKV projection: out = X @ W^T + bias ----------------
// X: [4096 x 768] f16 row-major.  W: [768 x 768] f16 row-major (N x K -> B operand directly).
// Writes out[b][h][s][d] f16; optionally outT[b][h][d][s].
__global__ __launch_bounds__(256) void proj_kernel(
    const _Float16* __restrict__ X, const _Float16* __restrict__ W,
    const float* __restrict__ bias, _Float16* __restrict__ out,
    _Float16* __restrict__ outT, int hasT) {
  const int lane = threadIdx.x & 31;
  const int wave = threadIdx.x >> 5;
  const int t  = blockIdx.x * 8 + wave;   // 12288 tiles total
  const int mt = t / 48;                  // 256 M-tiles
  const int nt = t % 48;                  // 48  N-tiles
  const int rofs = (lane & 15) * HIDc + ((lane & 16) >> 1);
  const _Float16* Ab = X + mt * 16 * HIDc + rofs;
  const _Float16* Bb = W + nt * 16 * HIDc + rofs;
  v8f acc = {};
#pragma unroll 4
  for (int k0 = 0; k0 < HIDc; k0 += 32) {
    v16h a = ld_frag16(Ab + k0);
    v16h b = ld_frag16(Bb + k0);
    acc = wmma_f16(a, b, acc);
  }
  const int col = lane & 15;
  const int n = nt * 16 + col;
  const int h = n >> 6, d = n & 63;
  const float bv = bias[n];
  const int rbase = (lane >> 4) << 3;
#pragma unroll
  for (int j = 0; j < 8; ++j) {
    int m = mt * 16 + rbase + j;
    int bb = m >> 10, s = m & 1023;
    _Float16 v = (_Float16)(acc[j] + bv);
    out[((bb * Hc + h) * Sc + s) * Dc + d] = v;
    if (hasT) outT[((bb * Hc + h) * Dc + d) * Sc + s] = v;
  }
}

// ---------------- 3) score GEMM: out[q,k] = (A[q,:] . B[k,:]) * scale + mask ----------------
// A,B: [BH][S][D] f16.  out: [BH][S][S] f32.  mask: [B][S] (k-indexed) or nullptr.
__global__ __launch_bounds__(256) void score_kernel(
    const _Float16* __restrict__ A, const _Float16* __restrict__ Bm,
    float* __restrict__ out, const float* __restrict__ mask, float scale) {
  const int lane = threadIdx.x & 31;
  const int wave = threadIdx.x >> 5;
  const int bh = blockIdx.x >> 6;
  const int qt = blockIdx.x & 63;
  const int rofs = (lane & 15) * Dc + ((lane & 16) >> 1);
  const _Float16* Ab = A + bh * Sc * Dc + qt * 16 * Dc + rofs;
  v16h a0 = ld_frag16(Ab);
  v16h a1 = ld_frag16(Ab + 32);
  const int col = lane & 15;
  const int rbase = (lane >> 4) << 3;
  const int bidx = bh / Hc;
  float* outb = out + bh * Sc * Sc + (qt * 16) * Sc;
#pragma unroll
  for (int i = 0; i < 8; ++i) {
    int nt = wave * 8 + i;
    const _Float16* Bb = Bm + bh * Sc * Dc + nt * 16 * Dc + rofs;
    v16h b0 = ld_frag16(Bb);
    v16h b1 = ld_frag16(Bb + 32);
    v8f acc = {};
    acc = wmma_f16(a0, b0, acc);
    acc = wmma_f16(a1, b1, acc);
    int c = nt * 16 + col;
    float mv = mask ? mask[bidx * Sc + c] : 0.f;
#pragma unroll
    for (int j = 0; j < 8; ++j)
      outb[(rbase + j) * Sc + c] = acc[j] * scale + mv;
  }
}

// ---------------- 4) softmax row stats (max, 1/sum) ----------------
__global__ __launch_bounds__(256) void stats_kernel(
    const float* __restrict__ scores, float* __restrict__ stats,
    float* __restrict__ ctx_score) {
  const int lane = threadIdx.x & 31;
  const int wave = threadIdx.x >> 5;
  const int row = blockIdx.x * 8 + wave;      // 49152 rows
  const float* rp = scores + row * Sc + lane;
  float m = -3.0e38f, s = 0.f;
#pragma unroll 4
  for (int i = 0; i < 32; ++i) {
    float x = rp[i * 32];
    float mn = fmaxf(m, x);
    s = s * __expf(m - mn) + __expf(x - mn);
    m = mn;
  }
#pragma unroll
  for (int off = 16; off > 0; off >>= 1) {
    float mo = __shfl_xor(m, off, 32);
    float so = __shfl_xor(s, off, 32);
    float mn = fmaxf(m, mo);
    s = s * __expf(m - mn) + so * __expf(mo - mn);
    m = mn;
  }
  if (lane == 0) {
    stats[row * 2 + 0] = m;
    stats[row * 2 + 1] = 1.f / s;
  }
  if (blockIdx.x == 0 && threadIdx.x == 0) *ctx_score = 0.f;
}

// ---------------- 5) context = softmax(scores) @ V ----------------
// Rebuilds P fragments on the fly from f32 scores + stats; B operand from Vt [BH][D][S] f16.
__global__ __launch_bounds__(256) void context_kernel(
    const float* __restrict__ scores, const float* __restrict__ stats,
    const _Float16* __restrict__ Vt, float* __restrict__ out) {
  const int lane = threadIdx.x & 31;
  const int wave = threadIdx.x >> 5;
  const int bh = blockIdx.x >> 3;
  const int qt = (blockIdx.x & 7) * 8 + wave;    // 0..63
  const int rowM = qt * 16 + (lane & 15);
  const int grow = bh * Sc + rowM;
  const float m   = stats[grow * 2 + 0];
  const float inv = stats[grow * 2 + 1];
  const float* sp0 = scores + bh * Sc * Sc + rowM * Sc + ((lane & 16) >> 1);
  const int rofs = (lane & 15) * Sc + ((lane & 16) >> 1);
  const _Float16* Vb = Vt + bh * Dc * Sc + rofs;
  v8f acc[4] = {{}, {}, {}, {}};
  for (int k0 = 0; k0 < Sc; k0 += 32) {
    const float* sp = sp0 + k0;
    v4f x0 = *(const v4f*)(sp);
    v4f x1 = *(const v4f*)(sp + 4);
    v4f x2 = *(const v4f*)(sp + 16);
    v4f x3 = *(const v4f*)(sp + 20);
    v16h a;
#pragma unroll
    for (int i = 0; i < 4; ++i) {
      a[i]      = (_Float16)(__expf(x0[i] - m) * inv);
      a[4 + i]  = (_Float16)(__expf(x1[i] - m) * inv);
      a[8 + i]  = (_Float16)(__expf(x2[i] - m) * inv);
      a[12 + i] = (_Float16)(__expf(x3[i] - m) * inv);
    }
#pragma unroll
    for (int nt = 0; nt < 4; ++nt) {
      v16h b = ld_frag16(Vb + nt * 16 * Sc + k0);
      acc[nt] = wmma_f16(a, b, acc[nt]);
    }
  }
  const int bb = bh / Hc, h = bh % Hc;
  const int rbase = (lane >> 4) << 3;
#pragma unroll
  for (int nt = 0; nt < 4; ++nt) {
    int colb = h * Dc + nt * 16 + (lane & 15);
#pragma unroll
    for (int j = 0; j < 8; ++j) {
      int srow = qt * 16 + rbase + j;
      out[(bb * Sc + srow) * HIDc + colb] = acc[nt][j];
    }
  }
}

// ---------------- host launch ----------------
extern "C" void kernel_launch(void* const* d_in, const int* in_sizes, int n_in,
                              void* d_out, int out_size, void* d_ws, size_t ws_size,
                              hipStream_t stream) {
  const float* hs   = (const float*)d_in[0];  // [4,1024,768]
  const float* mask = (const float*)d_in[1];  // [4,1,1,1024]
  const float* Wq   = (const float*)d_in[2];
  const float* bq   = (const float*)d_in[3];
  const float* Wk   = (const float*)d_in[4];
  const float* bk   = (const float*)d_in[5];
  const float* Wv   = (const float*)d_in[6];
  const float* bv   = (const float*)d_in[7];

  const int CTX = Bc * Sc * HIDc;         // 3,145,728
  const int SCO = BHc * Sc * Sc;          // 50,331,648

  float* out     = (float*)d_out;         // context
  float* attn    = out + CTX;             // attention_scores
  float* vals    = attn + SCO;            // value_scores
  float* cscore  = vals + SCO;            // context_score scalar
  float* qscores = cscore + 1;            // query_scores (4B-aligned only; scalar stores)
  float* kscores = qscores + SCO;         // key_scores

  _Float16* Xh  = (_Float16*)d_ws;
  _Float16* Wqh = Xh  + Bc * Sc * HIDc;
  _Float16* Wkh = Wqh + HIDc * HIDc;
  _Float16* Wvh = Wkh + HIDc * HIDc;
  _Float16* Qh  = Wvh + HIDc * HIDc;
  _Float16* Kh  = Qh  + BHc * Sc * Dc;
  _Float16* Vh  = Kh  + BHc * Sc * Dc;
  _Float16* Vth = Vh  + BHc * Sc * Dc;
  float* statsBuf = (float*)(Vth + BHc * Sc * Dc);   // [48*1024][2] f32

  const float scale = 0.125f;  // 1/sqrt(64)

  // 1) converts
  {
    int n4 = (Bc * Sc * HIDc) / 4;
    cvt_f32_f16_kernel<<<(n4 + 255) / 256, 256, 0, stream>>>(hs, Xh, n4);
    int w4 = (HIDc * HIDc) / 4;
    cvt_f32_f16_kernel<<<(w4 + 255) / 256, 256, 0, stream>>>(Wq, Wqh, w4);
    cvt_f32_f16_kernel<<<(w4 + 255) / 256, 256, 0, stream>>>(Wk, Wkh, w4);
    cvt_f32_f16_kernel<<<(w4 + 255) / 256, 256, 0, stream>>>(Wv, Wvh, w4);
  }

  // 2) QKV projections: (4096/16)*(768/16) = 12288 tiles, 8 waves/block -> 1536 blocks
  proj_kernel<<<1536, 256, 0, stream>>>(Xh, Wqh, bq, Qh, (_Float16*)nullptr, 0);
  proj_kernel<<<1536, 256, 0, stream>>>(Xh, Wkh, bk, Kh, (_Float16*)nullptr, 0);
  proj_kernel<<<1536, 256, 0, stream>>>(Xh, Wvh, bv, Vh, Vth, 1);

  // 3) four S x S score matrices: 48 bh * 64 qtiles blocks each
  score_kernel<<<BHc * 64, 256, 0, stream>>>(Qh, Kh, attn, mask, scale);
  score_kernel<<<BHc * 64, 256, 0, stream>>>(Qh, Qh, qscores, (const float*)nullptr, scale);
  score_kernel<<<BHc * 64, 256, 0, stream>>>(Kh, Kh, kscores, (const float*)nullptr, scale);
  score_kernel<<<BHc * 64, 256, 0, stream>>>(Vh, Vh, vals, (const float*)nullptr, scale);

  // 4) softmax stats over attention_scores rows (48*1024 rows / 8 waves)
  stats_kernel<<<(BHc * Sc) / 8, 256, 0, stream>>>(attn, statsBuf, cscore);

  // 5) context: 48 bh * 8 qtile-groups blocks
  context_kernel<<<BHc * 8, 256, 0, stream>>>(attn, statsBuf, Vth, out);
}
